// SFIModel_pipeline1_8478265442834
// MI455X (gfx1250) — matrix operand order repaired and checked by reading, not compiled, not run on hardware
//
#include <hip/hip_runtime.h>
#include <hip/hip_bf16.h>
#include <math.h>

// ---------------- problem constants ----------------
#define BATCH   64
#define NCDD    5
#define NHIS    50
#define LWORDS  30
#define EDIM    300
#define FDIM    150
#define EPAD    320       // E padded to 10 k-tiles of 32
#define FPAD    160       // F padded to 10 n-tiles of 16
#define KT      10        // k-tiles per tap
#define NT      10        // n-tiles
#define PADROW  3         // row padding for dilation <= 3
#define EROWS   38        // rows l = -3 .. +34
#define NTHREADS 160      // 5 waves of 32
#define INV_SCALE 0.05773502691896258f  // 1/sqrt(300)
#define LN_EPS  1e-5f

typedef __attribute__((ext_vector_type(16))) __bf16 v16bf;
typedef __attribute__((ext_vector_type(8)))  __bf16 v8bf;
typedef __attribute__((ext_vector_type(8)))  float  v8f;

// ---- LDS layout (bytes) for the encoder kernel ----
#define OFF_EMB  0                    // 38*320 bf16 = 24320 B ; reused as attn[32][160] f32 later
#define OFF_Y    24320                // 3*32*160 f32 = 61440 B
#define OFF_BIAS 85760                // 3*160 f32
#define OFF_LNG  87680                // 160 f32
#define OFF_LNB  88320
#define OFF_QL   88960
#define OFF_QW   89600
#define OFF_WDOT 90240                // 32 f32
#define OFF_TOK  90368                // 30 int
#define SMEM_BYTES 90496

// ---- workspace layout (bytes) ----
#define WS_WTILES   0                         // 900 tiles * 512 bf16 = 921600 B
#define WS_CDD      921600                    // 320*150 f32 = 192000 B
#define WS_HIS      (921600 + 192000)         // 3200*150 f32 = 1920000 B

// =====================================================================
// Kernel 0: swizzle conv weights (f32 [3][300][150] per branch) into
// bf16 WMMA B-fragment tiles: tile = (branch,tap,ktile,ntile),
// per-lane 16 contiguous bf16 (lane's B fragment for 16x16x32 bf16 WMMA).
// Lane L: n = ntile*16 + (L&15); element e -> k = ktile*32 + 16*(L>>4) + e.
// =====================================================================
__global__ __launch_bounds__(32)
void sfi_prep_weights(const float* __restrict__ w1,
                      const float* __restrict__ w2,
                      const float* __restrict__ w3,
                      __bf16* __restrict__ wtiles) {
    int tile = blockIdx.x;               // 0..899 = branch*300 + (tap*10+kt)*10 + nt
    int branch = tile / 300;
    int rem    = tile % 300;
    int tap    = rem / 100;
    int r2     = rem % 100;
    int kt     = r2 / 10;
    int nt     = r2 % 10;
    const float* w = (branch == 0) ? w1 : ((branch == 1) ? w2 : w3);

    int lane = threadIdx.x;              // 0..31
    int half = lane >> 4;
    int nl   = lane & 15;
    int n    = nt * 16 + nl;

    __bf16* out = wtiles + (size_t)tile * 512 + lane * 16;
#pragma unroll
    for (int e = 0; e < 16; ++e) {
        int k = kt * 32 + 16 * half + e;
        float v = 0.f;
        if (k < EDIM && n < FDIM)
            v = w[((size_t)tap * EDIM + k) * FDIM + n];
        out[e] = (__bf16)v;
    }
}

__device__ inline float quad_sum(float v) {
    v += __shfl_xor(v, 1);
    v += __shfl_xor(v, 2);
    return v;
}

// B fragment: 16 contiguous bf16 per lane (pre-swizzled in ws)
__device__ inline v16bf ld_fragB(const __bf16* p) {
    v8bf lo = *(const v8bf*)p;
    v8bf hi = *(const v8bf*)(p + 8);
    return __builtin_shufflevector(lo, hi,
        0,1,2,3,4,5,6,7,8,9,10,11,12,13,14,15);
}

// A fragment from LDS row: lane needs 16 contiguous bf16 at k-offset
// (16*half), matching the B fragment k order;
// caller passes p already offset by ktile*32 + 16*half.
__device__ inline v16bf ld_fragA(const __bf16* p) {
    v8bf lo = *(const v8bf*)p;
    v8bf hi = *(const v8bf*)(p + 8);
    return __builtin_shufflevector(lo, hi,
        0,1,2,3,4,5,6,7,8,9,10,11,12,13,14,15);
}

#define WMMA_BF16(a, b, c) __builtin_amdgcn_wmma_f32_16x16x32_bf16( \
        false, (a), false, (b), (short)0, (c), false, false)

// one K-step's fragments (SROA'd into 32 VGPRs)
struct Frags { v16bf b0, b1, a0, a1; };

// load fragments for K-step s (s = tap*10 + kt)
__device__ inline void load_step(Frags& fr, const __bf16* bbase,
                                 const __bf16* embL, int s,
                                 int ml, int half, int dil) {
    const int tap = (s >= 20) ? 2 : ((s >= 10) ? 1 : 0);
    const int kt  = s - tap * 10;
    const __bf16* bp = bbase + (size_t)s * (NT * 512);
    fr.b0 = ld_fragB(bp);
    fr.b1 = ld_fragB(bp + 512);
    const int row0 = ml + (tap - 1) * dil + PADROW;
    const __bf16* ar = embL + row0 * EPAD + kt * 32 + 16 * half;
    fr.a0 = ld_fragA(ar);
    fr.a1 = ld_fragA(ar + 16 * EPAD);
}

__device__ inline void mac_step(v8f& c00, v8f& c01, v8f& c10, v8f& c11,
                                const Frags& fr) {
    c00 = WMMA_BF16(fr.a0, fr.b0, c00);
    c01 = WMMA_BF16(fr.a0, fr.b1, c01);
    c10 = WMMA_BF16(fr.a1, fr.b0, c10);
    c11 = WMMA_BF16(fr.a1, fr.b1, c11);
}

// =====================================================================
// Kernel 1: news encoder. One workgroup per title.
// =====================================================================
__global__ __launch_bounds__(NTHREADS)
void sfi_encoder(const int*   __restrict__ titles,     // [n_titles][30]
                 const float* __restrict__ embedding,  // [V][300]
                 const __bf16* __restrict__ wtiles,    // swizzled B tiles
                 const float* __restrict__ b1,
                 const float* __restrict__ b2,
                 const float* __restrict__ b3,
                 const float* __restrict__ ln_g,
                 const float* __restrict__ ln_b,
                 const float* __restrict__ q_l,
                 const float* __restrict__ q_w,
                 float* __restrict__ reprs)             // [n_titles][150]
{
    extern __shared__ char smem[];
    __bf16* embL  = (__bf16*)(smem + OFF_EMB);
    float*  yL    = (float*)(smem + OFF_Y);
    float*  biasL = (float*)(smem + OFF_BIAS);
    float*  lngL  = (float*)(smem + OFF_LNG);
    float*  lnbL  = (float*)(smem + OFF_LNB);
    float*  qlL   = (float*)(smem + OFF_QL);
    float*  qwL   = (float*)(smem + OFF_QW);
    float*  wdotL = (float*)(smem + OFF_WDOT);
    int*    tokL  = (int*)(smem + OFF_TOK);

    const int T   = blockIdx.x;
    const int tid = threadIdx.x;

    // ---- stage token ids ----
    if (tid < LWORDS) tokL[tid] = titles[(size_t)T * LWORDS + tid];
    __syncthreads();

    // ---- phase A: params + gathered embedding (bf16, zero padded) ----
    {
        int f = tid;                                   // 0..159
        lngL[f] = (f < FDIM) ? ln_g[f] : 0.f;
        lnbL[f] = (f < FDIM) ? ln_b[f] : 0.f;
        qlL[f]  = (f < FDIM) ? q_l[f]  : 0.f;
        qwL[f]  = (f < FDIM) ? q_w[f]  : 0.f;
    }
    for (int i = tid; i < 3 * FPAD; i += NTHREADS) {
        int j = i / FPAD, f = i - j * FPAD;
        const float* bp = (j == 0) ? b1 : ((j == 1) ? b2 : b3);
        biasL[i] = (f < FDIM) ? bp[f] : 0.f;
    }
    for (int i = tid; i < EROWS * EPAD; i += NTHREADS) {
        int row = i / EPAD;            // 0..37 -> l = row-3
        int col = i - row * EPAD;
        int l   = row - PADROW;
        float v = 0.f;
        if (l >= 0 && l < LWORDS && col < EDIM)
            v = embedding[(size_t)tokL[l] * EDIM + col];
        embL[i] = (__bf16)v;
    }
    __syncthreads();

    // ---- phase B: dilated convs as WMMA GEMMs ----
    // Each wave owns 2x2 output tiles (2 M-tiles x 2 N-tiles, one branch):
    // per K-step: 2 B frags + 2 A frags -> 4 WMMAs.  Two-stage ping-pong
    // (no register rotation movs): iteration loads set1(s), MACs set0,
    // loads set0(s+1), MACs set1.
    {
        const int wave = tid >> 5;
        const int lane = tid & 31;
        const int half = lane >> 4;
        const int ml   = lane & 15;    // A/M and B/N lane index

        for (int g = wave; g < 15; g += 5) {     // 3 branches * 5 ntile-pairs
            const int branch = g / 5;
            const int npair  = g % 5;
            const int ntile0 = npair * 2;
            const int dil    = branch + 1;
            // B tile index = branch*300 + step*10 + ntile  (step = tap*10+kt)
            const __bf16* bbase = wtiles +
                (size_t)(branch * 300 + ntile0) * 512 + lane * 16;

            Frags f0, f1;
            load_step(f0, bbase, embL, 0, ml, half, dil);

            v8f acc00 = {}, acc01 = {}, acc10 = {}, acc11 = {};
#pragma unroll 1
            for (int i = 0; i < 14; ++i) {       // steps 0..27 MAC'd here
                const int s1 = 2 * i + 1;
                load_step(f1, bbase, embL, s1, ml, half, dil);
                mac_step(acc00, acc01, acc10, acc11, f0);
                load_step(f0, bbase, embL, s1 + 1, ml, half, dil);
                mac_step(acc00, acc01, acc10, acc11, f1);
            }
            // epilogue: steps 28, 29
            load_step(f1, bbase, embL, 29, ml, half, dil);
            mac_step(acc00, acc01, acc10, acc11, f0);
            mac_step(acc00, acc01, acc10, acc11, f1);

            // C/D layout: VGPR v -> M = 8*half + v (lanes 0-15) / +8 (16-31)
            const int n0 = ntile0 * 16 + ml;
            const float bias0 = biasL[branch * FPAD + n0];
            const float bias1 = biasL[branch * FPAD + n0 + 16];
            float* yb = yL + (size_t)branch * 32 * FPAD;
#pragma unroll
            for (int v = 0; v < 8; ++v) {
                const int m0 = 8 * half + v;
                const int m1 = m0 + 16;
                yb[(size_t)m0 * FPAD + n0]      = acc00[v] + bias0;
                yb[(size_t)m0 * FPAD + n0 + 16] = acc01[v] + bias1;
                yb[(size_t)m1 * FPAD + n0]      = acc10[v] + bias0;
                yb[(size_t)m1 * FPAD + n0 + 16] = acc11[v] + bias1;
            }
        }
    }
    __syncthreads();

    // ---- phase C: LayerNorm + ReLU + level attention (thread quads) ----
    float* attnL = (float*)(smem + OFF_EMB);   // reuse emb region: [32][160] f32
    if (tid < 4 * LWORDS) {
        const int l = tid >> 2;
        const int q = tid & 3;
        float qdot[3];
#pragma unroll
        for (int j = 0; j < 3; ++j) {
            float* yr = yL + ((size_t)j * 32 + l) * FPAD;
            float s = 0.f, s2 = 0.f;
            for (int f = q; f < FDIM; f += 4) { float v = yr[f]; s += v; s2 += v * v; }
            s  = quad_sum(s);
            s2 = quad_sum(s2);
            float mean = s * (1.f / FDIM);
            float var  = s2 * (1.f / FDIM) - mean * mean;
            float rstd = rsqrtf(var + LN_EPS);
            float qd = 0.f;
            for (int f = q; f < FDIM; f += 4) {
                float v = (yr[f] - mean) * rstd * lngL[f] + lnbL[f];
                v = fmaxf(v, 0.f);
                yr[f] = v;                       // in-place: y -> d
                qd += qlL[f] * v;
            }
            qdot[j] = quad_sum(qd);
        }
        // level softmax over 3
        float s0 = qdot[0] * INV_SCALE, s1 = qdot[1] * INV_SCALE, s2v = qdot[2] * INV_SCALE;
        float mx = fmaxf(s0, fmaxf(s1, s2v));
        float e0 = expf(s0 - mx), e1 = expf(s1 - mx), e2 = expf(s2v - mx);
        float inv = 1.f / (e0 + e1 + e2);
        e0 *= inv; e1 *= inv; e2 *= inv;
        const float* d0 = yL + (size_t)(0 * 32 + l) * FPAD;
        const float* d1 = yL + (size_t)(1 * 32 + l) * FPAD;
        const float* d2 = yL + (size_t)(2 * 32 + l) * FPAD;
        float wd = 0.f;
        for (int f = q; f < FDIM; f += 4) {
            float av = e0 * d0[f] + e1 * d1[f] + e2 * d2[f];
            attnL[(size_t)l * FPAD + f] = av;
            wd += qwL[f] * av;
        }
        wd = quad_sum(wd);
        if (q == 0) wdotL[l] = wd;
    }
    __syncthreads();

    // ---- phase D: word softmax + pooled representation ----
    if (tid < FDIM) {
        float mx = -3.4e38f;
        for (int l = 0; l < LWORDS; ++l) mx = fmaxf(mx, wdotL[l] * INV_SCALE);
        float se = 0.f, rep = 0.f;
        for (int l = 0; l < LWORDS; ++l) {
            float e = expf(wdotL[l] * INV_SCALE - mx);
            se  += e;
            rep += e * attnL[(size_t)l * FPAD + tid];
        }
        reprs[(size_t)T * FDIM + tid] = rep / se;
    }
}

// =====================================================================
// Kernel 2: scoring + log-softmax. One block per batch element.
// logits[c] = cdd[c] . (sum_h ltr_w[h] * his[h])  + ltr_b
// =====================================================================
__global__ __launch_bounds__(NTHREADS)
void sfi_score(const float* __restrict__ cddR,   // [64][5][150]
               const float* __restrict__ hisR,   // [64][50][150]
               const float* __restrict__ ltr_w,  // [1][50]
               const float* __restrict__ ltr_b,  // [1]
               float* __restrict__ out)          // [64][5]
{
    __shared__ float hs[FDIM];
    __shared__ float lg[NCDD];
    const int b = blockIdx.x;
    const int tid = threadIdx.x;

    if (tid < FDIM) {
        float s = 0.f;
        const float* hb = hisR + (size_t)b * NHIS * FDIM + tid;
        for (int h = 0; h < NHIS; ++h) s += ltr_w[h] * hb[(size_t)h * FDIM];
        hs[tid] = s;
    }
    __syncthreads();
    if (tid < NCDD) {
        float s = 0.f;
        const float* cb = cddR + ((size_t)b * NCDD + tid) * FDIM;
        for (int f = 0; f < FDIM; ++f) s += cb[f] * hs[f];
        lg[tid] = s + ltr_b[0];
    }
    __syncthreads();
    if (tid < NCDD) {
        float mx = -3.4e38f;
        for (int c = 0; c < NCDD; ++c) mx = fmaxf(mx, lg[c]);
        float se = 0.f;
        for (int c = 0; c < NCDD; ++c) se += expf(lg[c] - mx);
        out[(size_t)b * NCDD + tid] = lg[tid] - mx - logf(se);
    }
}

// =====================================================================
extern "C" void kernel_launch(void* const* d_in, const int* in_sizes, int n_in,
                              void* d_out, int out_size, void* d_ws, size_t ws_size,
                              hipStream_t stream) {
    const int*   cand     = (const int*)  d_in[0];   // [64][5][30]
    const int*   clicked  = (const int*)  d_in[1];   // [64][50][30]
    const float* emb      = (const float*)d_in[2];
    const float* w1       = (const float*)d_in[3];
    const float* b1       = (const float*)d_in[4];
    const float* w2       = (const float*)d_in[5];
    const float* b2       = (const float*)d_in[6];
    const float* w3       = (const float*)d_in[7];
    const float* b3       = (const float*)d_in[8];
    const float* ln_g     = (const float*)d_in[9];
    const float* ln_b     = (const float*)d_in[10];
    const float* q_l      = (const float*)d_in[11];
    const float* q_w      = (const float*)d_in[12];
    const float* ltr_w    = (const float*)d_in[13];
    const float* ltr_b    = (const float*)d_in[14];
    float* out = (float*)d_out;

    char* ws = (char*)d_ws;
    __bf16* wtiles = (__bf16*)(ws + WS_WTILES);
    float*  cddR   = (float*)(ws + WS_CDD);
    float*  hisR   = (float*)(ws + WS_HIS);

    sfi_prep_weights<<<900, 32, 0, stream>>>(w1, w2, w3, wtiles);

    sfi_encoder<<<BATCH * NCDD, NTHREADS, SMEM_BYTES, stream>>>(
        cand, emb, wtiles, b1, b2, b3, ln_g, ln_b, q_l, q_w, cddR);

    sfi_encoder<<<BATCH * NHIS, NTHREADS, SMEM_BYTES, stream>>>(
        clicked, emb, wtiles, b1, b2, b3, ln_g, ln_b, q_l, q_w, hisR);

    sfi_score<<<BATCH, NTHREADS, 0, stream>>>(cddR, hisR, ltr_w, ltr_b, out);
}
